// ACSCNN_29480655520372
// MI455X (gfx1250) — matrix operand order repaired and checked by reading, not compiled, not run on hardware
//
#include <hip/hip_runtime.h>
#include <hip/hip_bf16.h>

#define N_NODES 6890
#define N_ANG   8
#define K_CHEB  15
#define N_DESC  544
#define HIDDEN  64
#define MROWS   (N_ANG * N_NODES)   // 55120
#define DEG     16

typedef __attribute__((ext_vector_type(16))) __bf16 bf16x16;
typedef __attribute__((ext_vector_type(8)))  float  floatx8;

// ---------------------------------------------------------------------------
// Weight convert + transpose: W (slices, R, J) f32 -> WT (slices, J, R) bf16
// ---------------------------------------------------------------------------
__global__ void acs_wconv(const float* __restrict__ W, __bf16* __restrict__ WT,
                          int R, int J, long total) {
  long i = (long)blockIdx.x * blockDim.x + threadIdx.x;
  if (i >= total) return;
  int  j = (int)(i % J);
  long t = i / J;
  int  r = (int)(t % R);
  long s = t / R;
  WT[s * (long)J * R + (long)j * R + r] = (__bf16)W[i];
}

// ---------------------------------------------------------------------------
// T0 = tile(h, A): T[a*N + n, c] = h[n, c]
// ---------------------------------------------------------------------------
__global__ void acs_tile(const float* __restrict__ h, float* __restrict__ T, long NC) {
  long i = (long)blockIdx.x * blockDim.x + threadIdx.x;
  if (i < NC) {
    float v = h[i];
#pragma unroll
    for (int a = 0; a < N_ANG; ++a) T[(long)a * NC + i] = v;
  }
}

__global__ void acs_zero(float* __restrict__ p, long n) {
  long i = (long)blockIdx.x * blockDim.x + threadIdx.x;
  if (i < n) p[i] = 0.f;
}

// ---------------------------------------------------------------------------
// SpMM: one row per block, blockDim.x == C (multiple of 32).
// cheb==1: Tout[row,:] = 2*(L@Tin)[row,:] - Tprev[row,:]   (Tprev may == Tout)
// cheb==0: Tout[row,:] =   (L@Tin)[row,:]
// rows[i] == i/DEG by construction in the reference, so row index is implicit.
// ---------------------------------------------------------------------------
__global__ void acs_spmm(const int* __restrict__ cols, const float* __restrict__ vals,
                         const float* __restrict__ Tin, const float* __restrict__ Tprev,
                         float* __restrict__ Tout, int C, int cheb) {
  __shared__ int   scol[DEG];
  __shared__ float sval[DEG];
  int row = blockIdx.x;
  int c   = threadIdx.x;
  if (c < DEG) {
    scol[c] = cols[(size_t)row * DEG + c];
    sval[c] = vals[(size_t)row * DEG + c];
  }
  __syncthreads();
  float acc = 0.f;
#pragma unroll
  for (int j = 0; j < DEG; ++j)
    acc = fmaf(sval[j], Tin[(size_t)scol[j] * C + c], acc);
  size_t o = (size_t)row * C + c;
  Tout[o] = cheb ? (2.f * acc - Tprev[o]) : acc;
}

// ---------------------------------------------------------------------------
// WMMA GEMM: out(Nrows x Ncols) (+)= sum_ab A_ab(Nrows x Cdim) @ B_ab(Cdim x Ncols)
//   A: f32, physically (Ablocks, Nrows, Cdim) row-major (angle-stacked signal)
//   BT: bf16, (Ncols, Ablocks*Cdim) = B transposed; ldB = Ablocks*Cdim
// One wave computes a 16x64 strip: one A fragment (f32->bf16 in-register) and
// 4 pre-staged B fragments feed 4 back-to-back v_wmma_f32_16x16x32_bf16.
// Block = 128 threads = 4 waves = 64 rows x 64 cols of output.
// ---------------------------------------------------------------------------
__global__ __launch_bounds__(128)
void acs_wmma_gemm(const float* __restrict__ Amat, const __bf16* __restrict__ BT,
                   float* __restrict__ out, const float* __restrict__ bias,
                   int Nrows, int Ncols, int Cdim, int Ablocks,
                   int accum, int relu) {
  const int lane  = threadIdx.x & 31;
  const int wave  = threadIdx.x >> 5;
  const int hlf   = lane >> 4;      // 0 or 1 (wave half)
  const int ln    = lane & 15;
  const int rtile = blockIdx.x * 4 + wave;     // 16-row tile index
  const int col0  = blockIdx.y * 64;           // 64-col strip base
  const int ldB   = Ablocks * Cdim;

  int arow  = rtile * 16 + ln;                 // A-fragment row (M = lane&15)
  int arowc = arow < Nrows ? arow : 0;

  int bcol[4], bcolc[4];
#pragma unroll
  for (int jt = 0; jt < 4; ++jt) {
    bcol[jt]  = col0 + jt * 16 + ln;           // B-fragment col (N = lane&15)
    bcolc[jt] = bcol[jt] < Ncols ? bcol[jt] : 0;
  }

  floatx8 cacc[4];
#pragma unroll
  for (int jt = 0; jt < 4; ++jt)
    cacc[jt] = (floatx8){0.f, 0.f, 0.f, 0.f, 0.f, 0.f, 0.f, 0.f};

  for (int ab = 0; ab < Ablocks; ++ab) {
    const float*  abase = Amat + ((size_t)(ab * Nrows + arowc)) * Cdim;
    const __bf16* bb0 = BT + (size_t)bcolc[0] * ldB + (size_t)ab * Cdim;
    const __bf16* bb1 = BT + (size_t)bcolc[1] * ldB + (size_t)ab * Cdim;
    const __bf16* bb2 = BT + (size_t)bcolc[2] * ldB + (size_t)ab * Cdim;
    const __bf16* bb3 = BT + (size_t)bcolc[3] * ldB + (size_t)ab * Cdim;
    for (int cb = 0; cb < Cdim; cb += 32) {
      // Stage all 4 B fragments first (distinct regs -> one load clause,
      // then 4 back-to-back WMMAs; next iteration's loads overlap the WMMAs).
      bf16x16 bfr0 = *(const bf16x16*)(bb0 + cb + hlf * 16);
      bf16x16 bfr1 = *(const bf16x16*)(bb1 + cb + hlf * 16);
      bf16x16 bfr2 = *(const bf16x16*)(bb2 + cb + hlf * 16);
      bf16x16 bfr3 = *(const bf16x16*)(bb3 + cb + hlf * 16);
      // A fragment (16x32 bf16): lane half h holds K = {h*16..h*16+15}
      const float* ap = abase + cb + hlf * 16;
      float4 a0 = *(const float4*)(ap);
      float4 a1 = *(const float4*)(ap + 4);
      float4 a2 = *(const float4*)(ap + 8);
      float4 a3 = *(const float4*)(ap + 12);
      bf16x16 af;
      af[0]  = (__bf16)a0.x; af[1]  = (__bf16)a0.y; af[2]  = (__bf16)a0.z; af[3]  = (__bf16)a0.w;
      af[4]  = (__bf16)a1.x; af[5]  = (__bf16)a1.y; af[6]  = (__bf16)a1.z; af[7]  = (__bf16)a1.w;
      af[8]  = (__bf16)a2.x; af[9]  = (__bf16)a2.y; af[10] = (__bf16)a2.z; af[11] = (__bf16)a2.w;
      af[12] = (__bf16)a3.x; af[13] = (__bf16)a3.y; af[14] = (__bf16)a3.z; af[15] = (__bf16)a3.w;

      cacc[0] = __builtin_amdgcn_wmma_f32_16x16x32_bf16(false, af, false, bfr0,
                                                        (short)0, cacc[0], false, false);
      cacc[1] = __builtin_amdgcn_wmma_f32_16x16x32_bf16(false, af, false, bfr1,
                                                        (short)0, cacc[1], false, false);
      cacc[2] = __builtin_amdgcn_wmma_f32_16x16x32_bf16(false, af, false, bfr2,
                                                        (short)0, cacc[2], false, false);
      cacc[3] = __builtin_amdgcn_wmma_f32_16x16x32_bf16(false, af, false, bfr3,
                                                        (short)0, cacc[3], false, false);
    }
  }

  // Epilogue: D element e -> (M = hlf*8 + e, N = lane&15)
#pragma unroll
  for (int jt = 0; jt < 4; ++jt) {
    float bv = 0.f;
    if (bias != nullptr && bcol[jt] < Ncols) bv = bias[bcol[jt]];
#pragma unroll
    for (int e = 0; e < 8; ++e) {
      int row = rtile * 16 + hlf * 8 + e;
      if (row < Nrows && bcol[jt] < Ncols) {
        size_t idx = (size_t)row * Ncols + bcol[jt];
        float v = cacc[jt][e];
        if (accum) {
          out[idx] += v;
        } else {
          v += bv;
          if (relu) v = fmaxf(v, 0.f);
          out[idx] = v;
        }
      }
    }
  }
}

// ---------------------------------------------------------------------------
// BatchNorm1d (training-mode batch stats, biased var) + ReLU. One block/channel.
// ---------------------------------------------------------------------------
__global__ __launch_bounds__(256)
void acs_bn_relu(const float* __restrict__ acc, const float* __restrict__ bias,
                 const float* __restrict__ gamma, const float* __restrict__ beta,
                 float* __restrict__ out, int N, int C) {
  int c = blockIdx.x;
  int tid = threadIdx.x;
  __shared__ float ssum[256];
  __shared__ float ssq[256];
  float b = bias[c];
  float s = 0.f, q = 0.f;
  for (int n = tid; n < N; n += 256) {
    float v = acc[(size_t)n * C + c] + b;
    s += v; q += v * v;
  }
  ssum[tid] = s; ssq[tid] = q;
  __syncthreads();
  for (int st = 128; st > 0; st >>= 1) {
    if (tid < st) { ssum[tid] += ssum[tid + st]; ssq[tid] += ssq[tid + st]; }
    __syncthreads();
  }
  float mean = ssum[0] / (float)N;
  float var  = ssq[0] / (float)N - mean * mean;
  float sc   = gamma[c] * rsqrtf(var + 1e-5f);
  float bt   = beta[c];
  for (int n = tid; n < N; n += 256) {
    float v = (acc[(size_t)n * C + c] + b - mean) * sc + bt;
    out[(size_t)n * C + c] = fmaxf(v, 0.f);
  }
}

// ---------------------------------------------------------------------------
extern "C" void kernel_launch(void* const* d_in, const int* in_sizes, int n_in,
                              void* d_out, int out_size, void* d_ws, size_t ws_size,
                              hipStream_t stream) {
  (void)in_sizes; (void)n_in; (void)out_size; (void)ws_size;
  const float* x        = (const float*)d_in[0];
  // d_in[1] = rows: structurally repeat(arange(M), DEG) -> implicit in acs_spmm
  const int*   cols     = (const int*)d_in[2];
  const float* vals     = (const float*)d_in[3];
  const float* W1       = (const float*)d_in[4];
  const float* Wc       = (const float*)d_in[5];
  const float* conv_b   = (const float*)d_in[6];
  const float* bn_gamma = (const float*)d_in[7];
  const float* bn_beta  = (const float*)d_in[8];
  const float* fc2_w    = (const float*)d_in[9];
  const float* fc2_b    = (const float*)d_in[10];
  const float* fc3_w    = (const float*)d_in[11];
  const float* fc3_b    = (const float*)d_in[12];
  float* outp = (float*)d_out;

  const int N = N_NODES, M = MROWS, A = N_ANG;

  // Workspace carve (256B aligned)
  unsigned char* ws = (unsigned char*)d_ws;
  size_t off = 0;
  auto carve = [&](size_t bytes) -> void* {
    void* p = ws + off;
    off = (off + bytes + 255) & ~(size_t)255;
    return p;
  };
  float*  Ta   = (float*)carve((size_t)M * N_DESC * 4);
  float*  Tb   = (float*)carve((size_t)M * N_DESC * 4);
  float*  acc  = (float*)carve((size_t)N * HIDDEN * 4);
  float*  h1   = (float*)carve((size_t)N * HIDDEN * 4);
  float*  h2   = (float*)carve((size_t)N * 256 * 4);
  __bf16* W1T  = (__bf16*)carve((size_t)K_CHEB * (A * N_DESC) * HIDDEN * 2);
  __bf16* WcT  = (__bf16*)carve((size_t)5 * K_CHEB * (A * HIDDEN) * HIDDEN * 2);
  __bf16* fc2T = (__bf16*)carve((size_t)256 * HIDDEN * 2);
  __bf16* fc3T = (__bf16*)carve((size_t)N * 256 * 2);

  // ---- weight conversion (bf16, transposed to (J, R) per slice) ----
  {
    long t;
    t = (long)K_CHEB * (A * N_DESC) * HIDDEN;
    acs_wconv<<<dim3((t + 255) / 256), 256, 0, stream>>>(W1, W1T, A * N_DESC, HIDDEN, t);
    t = (long)5 * K_CHEB * (A * HIDDEN) * HIDDEN;
    acs_wconv<<<dim3((t + 255) / 256), 256, 0, stream>>>(Wc, WcT, A * HIDDEN, HIDDEN, t);
    t = (long)HIDDEN * 256;
    acs_wconv<<<dim3((t + 255) / 256), 256, 0, stream>>>(fc2_w, fc2T, HIDDEN, 256, t);
    t = (long)256 * N;
    acs_wconv<<<dim3((t + 255) / 256), 256, 0, stream>>>(fc3_w, fc3T, 256, N, t);
  }

  const float* hin = x;
  for (int layer = 0; layer < 6; ++layer) {
    const int C = (layer == 0) ? N_DESC : HIDDEN;
    const __bf16* WT = (layer == 0) ? W1T
                                    : (WcT + (size_t)(layer - 1) * K_CHEB * (A * HIDDEN) * HIDDEN);
    const size_t perK = (size_t)HIDDEN * (A * C);   // bf16 elements per Chebyshev slice

    // T0 = tile(hin, A)
    long nc = (long)N * C;
    acs_tile<<<dim3((nc + 255) / 256), 256, 0, stream>>>(hin, Ta, nc);
    // acc = 0
    long na = (long)N * HIDDEN;
    acs_zero<<<dim3((na + 255) / 256), 256, 0, stream>>>(acc, na);

    dim3 ggrid((N + 63) / 64, 1);   // Ncols = 64 -> one 64-col strip
    // out += proj(T0, W[0])
    acs_wmma_gemm<<<ggrid, 128, 0, stream>>>(Ta, WT, acc, nullptr, N, HIDDEN, C, A, 1, 0);
    // T1 = L @ T0
    acs_spmm<<<dim3(M), C, 0, stream>>>(cols, vals, Ta, nullptr, Tb, C, 0);
    acs_wmma_gemm<<<ggrid, 128, 0, stream>>>(Tb, WT + perK, acc, nullptr, N, HIDDEN, C, A, 1, 0);

    float* Tprev = Ta;  // T0
    float* Tcur  = Tb;  // T1
    for (int k = 2; k < K_CHEB; ++k) {
      // T2 = 2*(L@T1) - T0, written in place over T0 (row-local read-before-write)
      acs_spmm<<<dim3(M), C, 0, stream>>>(cols, vals, Tcur, Tprev, Tprev, C, 1);
      float* t = Tprev; Tprev = Tcur; Tcur = t;   // (Tprev,Tcur) = (T1,T2)
      acs_wmma_gemm<<<ggrid, 128, 0, stream>>>(Tcur, WT + (size_t)k * perK, acc, nullptr,
                                               N, HIDDEN, C, A, 1, 0);
    }
    // BN + ReLU -> h1
    acs_bn_relu<<<dim3(HIDDEN), 256, 0, stream>>>(acc, conv_b + layer * HIDDEN,
                                                  bn_gamma + layer * HIDDEN,
                                                  bn_beta + layer * HIDDEN, h1, N, HIDDEN);
    hin = h1;
  }

  // fc2: relu(h1 @ fc2_w + fc2_b) -> h2  (N x 256, K=64)
  acs_wmma_gemm<<<dim3((N + 63) / 64, (256 + 63) / 64), 128, 0, stream>>>(
      h1, fc2T, h2, fc2_b, N, 256, HIDDEN, 1, 0, 1);
  // fc3: h2 @ fc3_w + fc3_b -> out  (N x 6890, K=256)
  acs_wmma_gemm<<<dim3((N + 63) / 64, (N + 63) / 64), 128, 0, stream>>>(
      h2, fc3T, outp, fc3_b, N, N, 256, 1, 0, 0);
}
